// GGNN_27315992003012
// MI455X (gfx1250) — matrix operand order repaired and checked by reading, not compiled, not run on hardware
//
#include <hip/hip_runtime.h>
#include <math.h>

// GGNN on MI455X (gfx1250, wave32).
// Roofline: dominated by A (256MB fp32) streamed twice -> ~22us HBM floor.
// 16.4 GFLOP of matmul must run on v_wmma_f32_16x16x32_bf16 to stay
// memory-bound; A is converted fp32->bf16 in-register, s/weights are
// pre-converted bf16 in workspace (L2-resident).

#define DEV __device__ __forceinline__

typedef __attribute__((ext_vector_type(16))) __bf16 v16bf;
typedef __attribute__((ext_vector_type(8)))  float  v8f;

constexpr int B_  = 8;
constexpr int N_  = 1000;
constexpr int E_  = 4;
constexpr int D_  = 64;
constexpr int AD_ = 32;
constexpr int NE_ = N_ * E_;          // 4000
constexpr int TA_ = 2 * NE_;          // 8000 (A row length)
constexpr int NBLK = (N_ + 15) / 16;  // 63 row-blocks of 16
constexpr int KSPLIT = 5;             // k-dim split for big matmul
constexpr int KCHUNK = NE_ / KSPLIT;  // 800 (25 steps of 32)

// ---------- small helpers ----------

DEV unsigned short f2bfraw(float x) {
  unsigned u = __builtin_bit_cast(unsigned, x);
  unsigned r = u + 0x7FFFu + ((u >> 16) & 1u);   // round-to-nearest-even
  return (unsigned short)(r >> 16);
}
DEV __bf16 f2bf(float x) { return __builtin_bit_cast(__bf16, f2bfraw(x)); }

DEV v8f vzero8() {
  v8f r;
#pragma unroll
  for (int i = 0; i < 8; ++i) r[i] = 0.0f;
  return r;
}

struct BPack { uint4 lo, hi; };

// B-operand tile (32x16 bf16): lane l holds column n=l%16, 16 contiguous
// k-values starting at k0 + (l/16)*16.  p must already point at
// row (n) / col (k) for this lane.  Two b128 loads.
DEV v16bf load_btile(const unsigned short* p) {
  BPack t;
  t.lo = *(const uint4*)(p);
  t.hi = *(const uint4*)(p + 8);
  return __builtin_bit_cast(v16bf, t);
}

// A-operand tile (16x32) built from fp32 row data with in-register bf16
// conversion.  Lane l holds row m=l%16; k-groups: lanes<16 K={0..15},
// lanes>=16 K={16..31} (contiguous, matches B tile).  rowp points at this
// lane's (clamped) row; 'valid' zeroes out-of-range rows.
DEV v16bf atile_f32(const float* rowp, int kb, int half, bool valid) {
  const float4* g0 = (const float4*)(rowp + kb + half * 16);
  const float4* g1 = (const float4*)(rowp + kb + half * 16 + 8);
  float4 a0 = g0[0], a1 = g0[1], b0 = g1[0], b1 = g1[1];
  float f[16] = { a0.x, a0.y, a0.z, a0.w, a1.x, a1.y, a1.z, a1.w,
                  b0.x, b0.y, b0.z, b0.w, b1.x, b1.y, b1.z, b1.w };
  v16bf r;
#pragma unroll
  for (int i = 0; i < 16; ++i) r[i] = f2bf(valid ? f[i] : 0.0f);
  return r;
}

DEV v8f wmma_bf16(v16bf a, v16bf b, v8f c) {
  return __builtin_amdgcn_wmma_f32_16x16x32_bf16(
      /*neg_a=*/false, a, /*neg_b=*/false, b,
      /*c_mod=*/(short)0, c, /*reuse_a=*/false, /*reuse_b=*/false);
}

DEV uint4 pack8(v8f v) {  // 8 fp32 -> 8 packed bf16 (for b128 store)
  uint4 r;
  r.x = (unsigned)f2bfraw(v[0]) | ((unsigned)f2bfraw(v[1]) << 16);
  r.y = (unsigned)f2bfraw(v[2]) | ((unsigned)f2bfraw(v[3]) << 16);
  r.z = (unsigned)f2bfraw(v[4]) | ((unsigned)f2bfraw(v[5]) << 16);
  r.w = (unsigned)f2bfraw(v[6]) | ((unsigned)f2bfraw(v[7]) << 16);
  return r;
}

// ---------- kernels ----------

__global__ void k_cvt(const float* __restrict__ src,
                      unsigned short* __restrict__ dst, int n) {
  int i = blockIdx.x * blockDim.x + threadIdx.x;
  if (i < n) dst[i] = f2bfraw(src[i]);
}

__global__ void k_zero(float* __restrict__ p, int n) {
  int i = blockIdx.x * blockDim.x + threadIdx.x;
  int stride = gridDim.x * blockDim.x;
  for (; i < n; i += stride) p[i] = 0.0f;
}

// s_in[b,e,n,f] = sum_d state[b,n,d]*W_in[e,f,d]  -> stored transposed bf16
// sT[b][f][k], k = e*N + n.  One wave per (n-block, e, b).
__global__ void __launch_bounds__(32)
k_expert(const float* __restrict__ state,                 // (B,N,D) fp32
         const unsigned short* __restrict__ win_bf,       // (E,D,D) [e][f][d]
         const unsigned short* __restrict__ wout_bf,
         unsigned short* __restrict__ sT_in,              // (B,D,NE) [b][f][k]
         unsigned short* __restrict__ sT_out) {
  int lane = threadIdx.x;
  int nb = blockIdx.x, e = blockIdx.y, b = blockIdx.z;
  int n0 = nb * 16;
  int half = lane >> 4, l16 = lane & 15;
  int row = n0 + l16;
  bool rv = row < N_;
  const float* sp = state + ((size_t)b * N_ + (rv ? row : N_ - 1)) * D_;

  v8f acc_in[4], acc_out[4];
#pragma unroll
  for (int i = 0; i < 4; ++i) { acc_in[i] = vzero8(); acc_out[i] = vzero8(); }

#pragma unroll
  for (int ks = 0; ks < 2; ++ks) {                     // k over d: 2 steps of 32
    int kb = ks * 32;
    v16bf A = atile_f32(sp, kb, half, rv);
#pragma unroll
    for (int fb = 0; fb < 4; ++fb) {
      const unsigned short* bi =
          win_bf + ((size_t)e * D_ + fb * 16 + l16) * D_ + kb + half * 16;
      const unsigned short* bo =
          wout_bf + ((size_t)e * D_ + fb * 16 + l16) * D_ + kb + half * 16;
      acc_in[fb]  = wmma_bf16(A, load_btile(bi), acc_in[fb]);
      acc_out[fb] = wmma_bf16(A, load_btile(bo), acc_out[fb]);
    }
  }
  // C layout: lane's 8 elements are rows m = half*8 + v (consecutive k),
  // column f = fb*16 + l16.  Pack to bf16 and b128-store into sT[b][f][k].
  int krow0 = n0 + half * 8;
  if (krow0 + 8 <= N_) {
#pragma unroll
    for (int fb = 0; fb < 4; ++fb) {
      int f = fb * 16 + l16;
      size_t off = ((size_t)b * D_ + f) * NE_ + (size_t)e * N_ + krow0;
      *(uint4*)(sT_in + off)  = pack8(acc_in[fb]);
      *(uint4*)(sT_out + off) = pack8(acc_out[fb]);
    }
  }
}

// a_dir[b,i,f] += sum_{k in chunk} A[b,i,dir*NE+k] * sT_dir[b][f][k]
// One wave per (m-block, k-chunk, b*2+dir); fp32 atomics to accumulate.
__global__ void __launch_bounds__(32)
k_bigmm(const float* __restrict__ Aadj,                  // (B,N,8000) fp32
        const unsigned short* __restrict__ sT_in,
        const unsigned short* __restrict__ sT_out,
        float* __restrict__ a_in, float* __restrict__ a_out) {
  int lane = threadIdx.x;
  int mb = blockIdx.x;
  int kc = blockIdx.y;
  int bz = blockIdx.z;
  int b = bz >> 1, dir = bz & 1;
  const unsigned short* sT = dir ? sT_out : sT_in;
  float* dst = dir ? a_out : a_in;

  int half = lane >> 4, l16 = lane & 15;
  int row = mb * 16 + l16;
  bool rv = row < N_;
  const float* ap =
      Aadj + ((size_t)b * N_ + (rv ? row : N_ - 1)) * TA_ + (size_t)dir * NE_;

  v8f acc[4];
#pragma unroll
  for (int i = 0; i < 4; ++i) acc[i] = vzero8();

  int k0 = kc * KCHUNK;
  for (int ks = 0; ks < KCHUNK / 32; ++ks) {
    int kb = k0 + ks * 32;
    if (ks + 1 < KCHUNK / 32)
      __builtin_prefetch(ap + kb + 32, 0, 1);          // global_prefetch_b8
    v16bf A = atile_f32(ap, kb, half, rv);
#pragma unroll
    for (int fb = 0; fb < 4; ++fb) {
      const unsigned short* bp =
          sT + ((size_t)b * D_ + fb * 16 + l16) * NE_ + kb + half * 16;
      acc[fb] = wmma_bf16(A, load_btile(bp), acc[fb]);
    }
  }
#pragma unroll
  for (int fb = 0; fb < 4; ++fb) {
    int f = fb * 16 + l16;
#pragma unroll
    for (int v = 0; v < 8; ++v) {
      int m = mb * 16 + half * 8 + v;
      if (m < N_) atomicAdd(&dst[((size_t)b * N_ + m) * D_ + f], acc[fb][v]);
    }
  }
}

// GRU update: r,z = sigmoid([a_in,a_out,s] @ W.T); h = tanh([a_in,a_out,r*s] @ Wh.T)
// new = (1-z)*s + z*h.  One wave per (n-block, b); concat tile staged in LDS.
__global__ void __launch_bounds__(32)
k_gru(const float* __restrict__ a_in, const float* __restrict__ a_out,
      const float* __restrict__ state,
      const unsigned short* __restrict__ wr,
      const unsigned short* __restrict__ wz,
      const unsigned short* __restrict__ wh,
      float* __restrict__ out_state) {
  __shared__ float cat[16][200];   // 192 cols + pad
  __shared__ float st[16][68];
  __shared__ float rg[16][68];

  int lane = threadIdx.x;
  int nb = blockIdx.x, b = blockIdx.y;
  int n0 = nb * 16;
  int half = lane >> 4, l16 = lane & 15;

  for (int idx = lane; idx < 16 * 64; idx += 32) {
    int r = idx >> 6, c = idx & 63;
    int row = n0 + r;
    float ai = 0.f, ao = 0.f, s = 0.f;
    if (row < N_) {
      size_t o = ((size_t)b * N_ + row) * D_ + c;
      ai = a_in[o]; ao = a_out[o]; s = state[o];
    }
    cat[r][c] = ai; cat[r][64 + c] = ao; cat[r][128 + c] = s;
    st[r][c] = s;
  }
  __syncthreads();

  v8f r_acc[4], z_acc[4];
#pragma unroll
  for (int i = 0; i < 4; ++i) { r_acc[i] = vzero8(); z_acc[i] = vzero8(); }

#pragma unroll
  for (int ksi = 0; ksi < 6; ++ksi) {                  // k over 192: 6 steps
    int kb = ksi * 32;
    v16bf A = atile_f32(&cat[l16][0], kb, half, true);
#pragma unroll
    for (int fb = 0; fb < 4; ++fb) {
      const unsigned short* pr = wr + ((fb * 16 + l16) * 3 * D_) + kb + half * 16;
      const unsigned short* pz = wz + ((fb * 16 + l16) * 3 * D_) + kb + half * 16;
      r_acc[fb] = wmma_bf16(A, load_btile(pr), r_acc[fb]);
      z_acc[fb] = wmma_bf16(A, load_btile(pz), z_acc[fb]);
    }
  }
#pragma unroll
  for (int fb = 0; fb < 4; ++fb)
#pragma unroll
    for (int v = 0; v < 8; ++v) {
      r_acc[fb][v] = 1.0f / (1.0f + __expf(-r_acc[fb][v]));
      z_acc[fb][v] = 1.0f / (1.0f + __expf(-z_acc[fb][v]));
    }

  // scatter r into LDS (C layout -> natural layout), then gate the state slice
#pragma unroll
  for (int fb = 0; fb < 4; ++fb)
#pragma unroll
    for (int v = 0; v < 8; ++v)
      rg[half * 8 + v][fb * 16 + l16] = r_acc[fb][v];
  __syncthreads();
  for (int idx = lane; idx < 16 * 64; idx += 32) {
    int r = idx >> 6, c = idx & 63;
    cat[r][128 + c] = rg[r][c] * st[r][c];
  }
  __syncthreads();

  v8f h_acc[4];
#pragma unroll
  for (int i = 0; i < 4; ++i) h_acc[i] = vzero8();
#pragma unroll
  for (int ksi = 0; ksi < 6; ++ksi) {
    int kb = ksi * 32;
    v16bf A = atile_f32(&cat[l16][0], kb, half, true);
#pragma unroll
    for (int fb = 0; fb < 4; ++fb) {
      const unsigned short* ph = wh + ((fb * 16 + l16) * 3 * D_) + kb + half * 16;
      h_acc[fb] = wmma_bf16(A, load_btile(ph), h_acc[fb]);
    }
  }

#pragma unroll
  for (int fb = 0; fb < 4; ++fb) {
    int f = fb * 16 + l16;
#pragma unroll
    for (int v = 0; v < 8; ++v) {
      int m = half * 8 + v;
      float h = tanhf(h_acc[fb][v]);
      float z = z_acc[fb][v];
      float s = st[m][f];
      float ns = (1.0f - z) * s + z * h;
      int row = n0 + m;
      if (row < N_) out_state[((size_t)b * N_ + row) * D_ + f] = ns;
    }
  }
}

// out[b,i] = sum_f tanh( [s2,ann] @ Wo1.T )[i,f] * Wo2[f]
__global__ void __launch_bounds__(32)
k_out(const float* __restrict__ s2, const float* __restrict__ ann,
      const unsigned short* __restrict__ wo1,            // (D, D+AD) [f][c]
      const float* __restrict__ wo2,                     // (1, D)
      float* __restrict__ out) {
  int lane = threadIdx.x;
  int nb = blockIdx.x, b = blockIdx.y;
  int half = lane >> 4, l16 = lane & 15;
  int row = nb * 16 + l16;
  bool rv = row < N_;
  const float* sp = s2  + ((size_t)b * N_ + (rv ? row : N_ - 1)) * D_;
  const float* ap = ann + ((size_t)b * N_ + (rv ? row : N_ - 1)) * AD_;

  v8f acc[4];
#pragma unroll
  for (int i = 0; i < 4; ++i) acc[i] = vzero8();

#pragma unroll
  for (int ksi = 0; ksi < 3; ++ksi) {                   // k over 96: 3 steps
    int kb = ksi * 32;
    v16bf A = (ksi < 2) ? atile_f32(sp, kb, half, rv)
                        : atile_f32(ap, 0, half, rv);
#pragma unroll
    for (int fb = 0; fb < 4; ++fb) {
      const unsigned short* bp =
          wo1 + ((fb * 16 + l16) * (D_ + AD_)) + kb + half * 16;
      acc[fb] = wmma_bf16(A, load_btile(bp), acc[fb]);
    }
  }

  float partial[8];
#pragma unroll
  for (int v = 0; v < 8; ++v) partial[v] = 0.0f;
#pragma unroll
  for (int fb = 0; fb < 4; ++fb) {
    float w = wo2[fb * 16 + l16];
#pragma unroll
    for (int v = 0; v < 8; ++v) partial[v] += tanhf(acc[fb][v]) * w;
  }
  // reduce over the 16 lanes sharing a row (both halves independently)
#pragma unroll
  for (int v = 0; v < 8; ++v)
#pragma unroll
    for (int m = 1; m < 16; m <<= 1)
      partial[v] += __shfl_xor(partial[v], m, 16);

  if (l16 == 0) {
#pragma unroll
    for (int v = 0; v < 8; ++v) {
      int r = nb * 16 + half * 8 + v;
      if (r < N_) out[(size_t)b * N_ + r] = partial[v];
    }
  }
}

// ---------- host ----------

extern "C" void kernel_launch(void* const* d_in, const int* in_sizes, int n_in,
                              void* d_out, int out_size, void* d_ws,
                              size_t ws_size, hipStream_t stream) {
  const float* prop_state = (const float*)d_in[0];
  const float* annotation = (const float*)d_in[1];
  const float* Aadj       = (const float*)d_in[2];
  const float* W_in       = (const float*)d_in[3];
  const float* W_out      = (const float*)d_in[4];
  const float* W_r        = (const float*)d_in[5];
  const float* W_z        = (const float*)d_in[6];
  const float* W_h        = (const float*)d_in[7];
  const float* Wo1        = (const float*)d_in[8];
  const float* Wo2        = (const float*)d_in[9];
  float* out = (float*)d_out;

  char* w = (char*)d_ws;
  size_t off = 0;
  auto take = [&](size_t bytes) -> char* {
    char* p = w + off;
    off = (off + bytes + 255) & ~(size_t)255;
    return p;
  };
  float* s1     = (float*)take((size_t)B_ * N_ * D_ * 4);
  float* s2     = (float*)take((size_t)B_ * N_ * D_ * 4);
  float* a_in   = (float*)take((size_t)B_ * N_ * D_ * 4);
  float* a_out  = (float*)take((size_t)B_ * N_ * D_ * 4);
  unsigned short* sT_in  = (unsigned short*)take((size_t)B_ * D_ * NE_ * 2);
  unsigned short* sT_out = (unsigned short*)take((size_t)B_ * D_ * NE_ * 2);
  unsigned short* wr_bf  = (unsigned short*)take((size_t)D_ * 3 * D_ * 2);
  unsigned short* wz_bf  = (unsigned short*)take((size_t)D_ * 3 * D_ * 2);
  unsigned short* wh_bf  = (unsigned short*)take((size_t)D_ * 3 * D_ * 2);
  unsigned short* wo1_bf = (unsigned short*)take((size_t)D_ * (D_ + AD_) * 2);
  unsigned short* win_bf = (unsigned short*)take((size_t)E_ * D_ * D_ * 2);
  unsigned short* wout_bf = (unsigned short*)take((size_t)E_ * D_ * D_ * 2);
  if (off > ws_size) return;  // workspace too small

  // 1) weight conversions (bf16, layouts already B-operand friendly)
  auto cvt = [&](const float* s, unsigned short* d, int n) {
    k_cvt<<<(n + 255) / 256, 256, 0, stream>>>(s, d, n);
  };
  cvt(W_r, wr_bf, D_ * 3 * D_);
  cvt(W_z, wz_bf, D_ * 3 * D_);
  cvt(W_h, wh_bf, D_ * 3 * D_);
  cvt(Wo1, wo1_bf, D_ * (D_ + AD_));
  cvt(W_in, win_bf, E_ * D_ * D_);
  cvt(W_out, wout_bf, E_ * D_ * D_);

  // 2) two propagation rounds
  for (int prop = 0; prop < 2; ++prop) {
    const float* sin  = (prop == 0) ? prop_state : s1;
    float*       snew = (prop == 0) ? s1 : s2;
    k_zero<<<512, 256, 0, stream>>>(a_in,  B_ * N_ * D_);
    k_zero<<<512, 256, 0, stream>>>(a_out, B_ * N_ * D_);
    k_expert<<<dim3(NBLK, E_, B_), 32, 0, stream>>>(
        sin, win_bf, wout_bf, sT_in, sT_out);
    k_bigmm<<<dim3(NBLK, KSPLIT, B_ * 2), 32, 0, stream>>>(
        Aadj, sT_in, sT_out, a_in, a_out);
    k_gru<<<dim3(NBLK, B_), 32, 0, stream>>>(
        a_in, a_out, sin, wr_bf, wz_bf, wh_bf, snew);
  }

  // 3) output head
  k_out<<<dim3(NBLK, B_), 32, 0, stream>>>(s2, annotation, wo1_bf, Wo2, out);
}
